// AudioEncoder_17291538333981
// MI455X (gfx1250) — matrix operand order repaired and checked by reading, not compiled, not run on hardware
//
#include <hip/hip_runtime.h>
#include <hip/hip_bf16.h>

// ---------------------------------------------------------------------------
// Types for CDNA5 WMMA (gfx1250, wave32)
// ---------------------------------------------------------------------------
typedef __attribute__((ext_vector_type(16))) __bf16 bf16x16;  // A/B fragment (8 VGPRs)
typedef __attribute__((ext_vector_type(8)))  float  f32x8;    // C/D fragment (8 VGPRs)

__device__ __forceinline__ f32x8 wmma_bf16(bf16x16 a, bf16x16 b, f32x8 c) {
  // D = A(16x32 bf16) * B(32x16 bf16) + C(16x16 f32)
  return __builtin_amdgcn_wmma_f32_16x16x32_bf16(false, a, false, b, (short)0, c,
                                                 false, false);
}

// Load an A-fragment: elements 0..7 from p0 (16B), elements 8..15 from p1 (16B).
__device__ __forceinline__ bf16x16 load_frag_2x16B(const __bf16* p0, const __bf16* p1) {
  bf16x16 r;
  *(uint4*)&r       = *(const uint4*)p0;
  *((uint4*)&r + 1) = *(const uint4*)p1;
  return r;
}
// Load a B-fragment: elements 0..7 from p (16B), elements 8..15 from p+16 (16B).
__device__ __forceinline__ bf16x16 load_frag_32B(const __bf16* p) {
  bf16x16 r;
  *(uint4*)&r       = *(const uint4*)p;
  *((uint4*)&r + 1) = *(const uint4*)(p + 16);
  return r;
}

// ---------------------------------------------------------------------------
// f32 -> bf16 conversion, 4 elements per thread (n must be a multiple of 4)
// ---------------------------------------------------------------------------
__global__ void f32_to_bf16_kernel(const float* __restrict__ src,
                                   __bf16* __restrict__ dst, int n) {
  const int i4 = (blockIdx.x * blockDim.x + threadIdx.x) * 4;
  if (i4 < n) {
    const float4 f = *(const float4*)(src + i4);
    __bf16 o[4] = {(__bf16)f.x, (__bf16)f.y, (__bf16)f.z, (__bf16)f.w};
    *(uint2*)(dst + i4) = *(uint2*)o;
  }
}

// ---------------------------------------------------------------------------
// Row LayerNorm over C=1024: wave32 shuffle reduction + 8-partial LDS combine.
// ---------------------------------------------------------------------------
__global__ __launch_bounds__(256) void layernorm_kernel(
    const float* __restrict__ x, const float* __restrict__ g,
    const float* __restrict__ b, float* __restrict__ outF,
    __bf16* __restrict__ outBF, int C_) {
  __shared__ float part1[8];
  __shared__ float part2[8];
  const int row  = blockIdx.x;
  const int tid  = threadIdx.x;
  const int lane = tid & 31, wid = tid >> 5;
  const float* xr = x + (size_t)row * C_;

  float v[4];
  float s = 0.f;
#pragma unroll
  for (int i = 0; i < 4; ++i) { v[i] = xr[tid + 256 * i]; s += v[i]; }
#pragma unroll
  for (int off = 16; off > 0; off >>= 1) s += __shfl_xor(s, off, 32);
  if (lane == 0) part1[wid] = s;
  __syncthreads();
  float tot = 0.f;
#pragma unroll
  for (int w = 0; w < 8; ++w) tot += part1[w];
  const float mean = tot * (1.f / 1024.f);

  s = 0.f;
#pragma unroll
  for (int i = 0; i < 4; ++i) { const float d = v[i] - mean; s += d * d; }
#pragma unroll
  for (int off = 16; off > 0; off >>= 1) s += __shfl_xor(s, off, 32);
  if (lane == 0) part2[wid] = s;
  __syncthreads();
  tot = 0.f;
#pragma unroll
  for (int w = 0; w < 8; ++w) tot += part2[w];
  const float rstd = rsqrtf(tot * (1.f / 1024.f) + 1e-5f);

#pragma unroll
  for (int i = 0; i < 4; ++i) {
    const int c = tid + 256 * i;
    const float y = (v[i] - mean) * rstd * g[c] + b[c];
    const size_t idx = (size_t)row * C_ + c;
    if (outF)  outF[idx]  = y;
    if (outBF) outBF[idx] = (__bf16)y;
  }
}

// ---------------------------------------------------------------------------
// qu = q + pos_bias_u[h,d], qv = q + pos_bias_v[h,d]  (bf16 outputs)
// ---------------------------------------------------------------------------
__global__ void make_quv_kernel(const float* __restrict__ q,
                                const float* __restrict__ u,
                                const float* __restrict__ v,
                                __bf16* __restrict__ qu,
                                __bf16* __restrict__ qv, int total) {
  const int i4 = (blockIdx.x * blockDim.x + threadIdx.x) * 4;
  if (i4 < total) {
    const int c = i4 & 1023;  // C = 1024
    const float4 qf = *(const float4*)(q + i4);
    const float4 uf = *(const float4*)(u + c);
    const float4 vf = *(const float4*)(v + c);
    __bf16 ou[4] = {(__bf16)(qf.x + uf.x), (__bf16)(qf.y + uf.y),
                    (__bf16)(qf.z + uf.z), (__bf16)(qf.w + uf.w)};
    __bf16 ov[4] = {(__bf16)(qf.x + vf.x), (__bf16)(qf.y + vf.y),
                    (__bf16)(qf.z + vf.z), (__bf16)(qf.w + vf.w)};
    *(uint2*)(qu + i4) = *(uint2*)ou;
    *(uint2*)(qv + i4) = *(uint2*)ov;
  }
}

// ---------------------------------------------------------------------------
// WMMA GEMM, register-staged software pipeline:
//   store tile t (regs->LDS) ; issue global loads for tile t+1 into same regs ;
//   barrier ; 16 WMMAs on tile t while loads fly.
// Block: 256 threads (8 waves). Tile 128x128, K-tile 64 (2 WMMA k-steps).
// Wave grid 4(M) x 2(N): each wave computes 32x64 = 2x4 fragments.
// Requires M%128==0, N%128==0, K%64==0.
// ---------------------------------------------------------------------------
template <bool HAS_BIAS, bool RELU, bool HAS_RES, bool OUT_F, bool OUT_BF>
__global__ __launch_bounds__(256) void gemm_bf16_wmma(
    const __bf16* __restrict__ A, const __bf16* __restrict__ Bw,
    const float* __restrict__ bias, const float* __restrict__ residual,
    float* __restrict__ outF, __bf16* __restrict__ outBF,
    int M, int N, int K) {
  __shared__ __align__(16) __bf16 sA[128][72];   // [m][k] (pad 8)
  __shared__ __align__(16) __bf16 sBt[128][72];  // [n][k] (B transposed, pad 8)

  const int tid  = threadIdx.x;
  const int lane = tid & 31;
  const int wid  = tid >> 5;
  const int l15  = lane & 15;
  const int hi   = lane >> 4;      // 0 or 1 (half-wave)
  const int wm   = wid & 3;        // 4 waves along M (32 rows each)
  const int wn   = wid >> 2;       // 2 waves along N (64 cols each)
  const int m0   = blockIdx.y * 128;
  const int n0   = blockIdx.x * 128;

  // Loader geometry
  const int arow = tid >> 1;             // A: row 0..127
  const int akof = (tid & 1) * 32;       // A: k offset 0/32
  const int bk2  = (tid & 31) * 2;       // B: k-row pair base (0,2,..,62)
  const int bcol = (tid >> 5) * 16;      // B: 16-column segment base

  f32x8 acc[2][4];
#pragma unroll
  for (int mi = 0; mi < 2; ++mi)
#pragma unroll
    for (int nj = 0; nj < 4; ++nj) acc[mi][nj] = (f32x8)(0.f);

  uint4 aR[4];        // 32 bf16 of the A tile
  uint  bl[8], bh[8]; // 16 cols x 2 k-rows of the B tile (2 bf16 per uint)

  auto load_tile = [&](int kt) {
    const __bf16* asrc = A + (size_t)(m0 + arow) * K + kt + akof;
#pragma unroll
    for (int q = 0; q < 4; ++q) aR[q] = ((const uint4*)asrc)[q];
    const __bf16* b0 = Bw + (size_t)(kt + bk2) * N + n0 + bcol;
    const __bf16* b1 = b0 + N;
    const uint4 t0 = ((const uint4*)b0)[0], t1 = ((const uint4*)b0)[1];
    const uint4 u0 = ((const uint4*)b1)[0], u1 = ((const uint4*)b1)[1];
    bl[0] = t0.x; bl[1] = t0.y; bl[2] = t0.z; bl[3] = t0.w;
    bl[4] = t1.x; bl[5] = t1.y; bl[6] = t1.z; bl[7] = t1.w;
    bh[0] = u0.x; bh[1] = u0.y; bh[2] = u0.z; bh[3] = u0.w;
    bh[4] = u1.x; bh[5] = u1.y; bh[6] = u1.z; bh[7] = u1.w;
  };
  auto store_tile = [&]() {
#pragma unroll
    for (int q = 0; q < 4; ++q) *(uint4*)&sA[arow][akof + 8 * q] = aR[q];
#pragma unroll
    for (int i = 0; i < 8; ++i) {  // uint i holds cols 2i,2i+1 of rows bk2,bk2+1
      const uint lo = bl[i], hw = bh[i];
      *(uint*)&sBt[bcol + 2 * i][bk2]     = (lo & 0xffffu) | (hw << 16);
      *(uint*)&sBt[bcol + 2 * i + 1][bk2] = (lo >> 16) | (hw & 0xffff0000u);
    }
  };

  load_tile(0);
  for (int kt = 0; kt < K; kt += 64) {
    __syncthreads();           // previous compute done reading LDS
    store_tile();              // ds_store consumes regs at issue ...
    if (kt + 64 < K) load_tile(kt + 64);  // ... so refill them immediately
    __syncthreads();           // tile visible; next-tile loads are in flight

#pragma unroll
    for (int ks = 0; ks < 2; ++ks) {
      const int kb = ks * 32;
      bf16x16 af[2];
#pragma unroll
      for (int mi = 0; mi < 2; ++mi) {
        const int row = wm * 32 + mi * 16 + l15;
        // ISA A layout: lane<16 -> K 0-7/16-23; lane>=16 -> K 8-15/24-31
        af[mi] = load_frag_2x16B(&sA[row][kb + hi * 8], &sA[row][kb + hi * 8 + 16]);
      }
#pragma unroll
      for (int nj = 0; nj < 4; ++nj) {
        const int col = wn * 64 + nj * 16 + l15;
        const bf16x16 bf_ = load_frag_32B(&sBt[col][kb + hi * 8]);
#pragma unroll
        for (int mi = 0; mi < 2; ++mi)
          acc[mi][nj] = wmma_bf16(af[mi], bf_, acc[mi][nj]);
      }
    }
  }

#pragma unroll
  for (int nj = 0; nj < 4; ++nj) {
    const int col = n0 + wn * 64 + nj * 16 + l15;
    const float bcolv = HAS_BIAS ? bias[col] : 0.f;
#pragma unroll
    for (int mi = 0; mi < 2; ++mi)
#pragma unroll
      for (int r = 0; r < 8; ++r) {
        const int row = m0 + wm * 32 + mi * 16 + r + 8 * hi;  // C frag: M = r + 8*hi
        float val = acc[mi][nj][r] + bcolv;
        if (RELU) val = fmaxf(val, 0.f);
        const size_t idx = (size_t)row * N + col;
        if (HAS_RES) val += residual[idx];
        if (OUT_F)  outF[idx]  = val;
        if (OUT_BF) outBF[idx] = (__bf16)val;
      }
  }
}

// ---------------------------------------------------------------------------
// Flash-style relative-position attention (Transformer-XL form):
//   S = ((q+u)K^T + (q+v)P^T) * D^-0.5 ; softmax over keys (mask); O = S V
// Grid: B*H*(N/128) blocks, 256 threads (8 waves); each wave owns 16 query
// rows. K/V/P tiles of 32 keys staged in LDS with register-pipelined loads;
// scores never leave the WGP.
// ---------------------------------------------------------------------------
__global__ __launch_bounds__(256) void attention_kernel(
    const __bf16* __restrict__ qu, const __bf16* __restrict__ qv,
    const __bf16* __restrict__ kk, const __bf16* __restrict__ vv,
    const __bf16* __restrict__ pp, const unsigned char* __restrict__ mask,
    __bf16* __restrict__ outO) {
  constexpr int Nn = 1024, Cc = 1024;
  __shared__ __align__(16) __bf16 sK[32][72];          // [m][d]
  __shared__ __align__(16) __bf16 sP[32][72];          // [m][d]
  __shared__ __align__(16) __bf16 sVt[64][40];         // [d][m] (transposed)
  __shared__ __align__(16) __bf16 sProb[8][16][32];    // per-wave prob bounce

  const int tid  = threadIdx.x;
  const int lane = tid & 31;
  const int wid  = tid >> 5;
  const int l15  = lane & 15;
  const int hi   = lane >> 4;

  const int blk = blockIdx.x;            // b*H*8 + h*8 + qb
  const int qb  = blk & 7;
  const int h   = (blk >> 3) & 15;
  const int b   = blk >> 7;
  const int n0  = qb * 128 + wid * 16;   // this wave's 16 query rows

  // Preload this wave's Q fragments (q+u and q+v), d split into two K-steps.
  bf16x16 au[2], av[2];
  {
    const size_t qrow = ((size_t)b * Nn + n0 + l15) * Cc + h * 64;
#pragma unroll
    for (int t = 0; t < 2; ++t) {
      const int kb = t * 32 + hi * 8;
      au[t] = load_frag_2x16B(qu + qrow + kb, qu + qrow + kb + 16);
      av[t] = load_frag_2x16B(qv + qrow + kb, qv + qrow + kb + 16);
    }
  }

  f32x8 o[4];
#pragma unroll
  for (int jd = 0; jd < 4; ++jd) o[jd] = (f32x8)(0.f);
  float mrow[8], lrow[8];
#pragma unroll
  for (int r = 0; r < 8; ++r) { mrow[r] = -1e30f; lrow[r] = 0.f; }
  const float scale = 0.125f;  // D^-0.5, D = 64

  // Tile loader: 32 keys x 64 dims each for K/P/V, 16B per thread per tensor.
  const int lrow_ = tid >> 3;            // key row 0..31
  const int ldof  = (tid & 7) * 8;       // dim offset 0..56
  uint4 kR, pR, vR;
  auto load_tiles = [&](int m0_) {
    const size_t krow = ((size_t)b * Nn + m0_ + lrow_) * Cc + h * 64 + ldof;
    kR = *(const uint4*)(kk + krow);
    vR = *(const uint4*)(vv + krow);
    const size_t prow = ((size_t)(m0_ + lrow_)) * Cc + h * 64 + ldof;
    pR = *(const uint4*)(pp + prow);
  };
  auto store_tiles = [&]() {
    *(uint4*)&sK[lrow_][ldof] = kR;
    *(uint4*)&sP[lrow_][ldof] = pR;
    __bf16 vtmp[8];
    *(uint4*)vtmp = vR;
#pragma unroll
    for (int j = 0; j < 8; ++j) sVt[ldof + j][lrow_] = vtmp[j];  // transpose
  };

  load_tiles(0);
  for (int m0 = 0; m0 < Nn; m0 += 32) {
    __syncthreads();                     // previous compute done reading LDS
    store_tiles();
    if (m0 + 32 < Nn) load_tiles(m0 + 32);  // overlap next loads with compute
    __syncthreads();

    // Scores: two 16x16 column fragments covering the 32 keys.
    f32x8 s[2];
#pragma unroll
    for (int j = 0; j < 2; ++j) {
      f32x8 accs = (f32x8)(0.f);
#pragma unroll
      for (int t = 0; t < 2; ++t) {
        const bf16x16 bk = load_frag_32B(&sK[j * 16 + l15][t * 32 + hi * 8]);
        accs = wmma_bf16(au[t], bk, accs);                     // (q+u) . k
        const bf16x16 bp = load_frag_32B(&sP[j * 16 + l15][t * 32 + hi * 8]);
        accs = wmma_bf16(av[t], bp, accs);                     // (q+v) . p
      }
      const int mcol = m0 + j * 16 + l15;
      const bool ok  = mask[(size_t)b * Nn + mcol] != 0;
#pragma unroll
      for (int r = 0; r < 8; ++r) {
        const float x = accs[r] * scale;
        s[j][r] = ok ? x : -1e30f;
      }
    }

    // Online softmax per query row (row lives across 16 lanes of a half-wave).
#pragma unroll
    for (int r = 0; r < 8; ++r) {
      float cur = fmaxf(s[0][r], s[1][r]);
#pragma unroll
      for (int off = 1; off < 16; off <<= 1)
        cur = fmaxf(cur, __shfl_xor(cur, off, 16));
      const float mn    = fmaxf(mrow[r], cur);
      const float alpha = __expf(mrow[r] - mn);
      const float p0 = __expf(s[0][r] - mn);
      const float p1 = __expf(s[1][r] - mn);
      float rs = p0 + p1;
#pragma unroll
      for (int off = 1; off < 16; off <<= 1) rs += __shfl_xor(rs, off, 16);
      lrow[r] = lrow[r] * alpha + rs;
      mrow[r] = mn;
#pragma unroll
      for (int jd = 0; jd < 4; ++jd) o[jd][r] *= alpha;
      // C-frag (M=r+8*hi, N=l15) -> LDS, to be re-read in A-frag layout
      sProb[wid][r + 8 * hi][l15]      = (__bf16)p0;
      sProb[wid][r + 8 * hi][16 + l15] = (__bf16)p1;
    }
    // Intra-wave LDS RAW: DS ops are in-order per wave; just drain DScnt.
    asm volatile("s_wait_dscnt 0" ::: "memory");

    const bf16x16 ap = load_frag_2x16B(&sProb[wid][l15][8 * hi],
                                       &sProb[wid][l15][8 * hi + 16]);
#pragma unroll
    for (int jd = 0; jd < 4; ++jd) {
      const bf16x16 bv_ = load_frag_32B(&sVt[jd * 16 + l15][hi * 8]);
      o[jd] = wmma_bf16(ap, bv_, o[jd]);  // O += P~ . V
    }
  }

  // Normalize and write O in (B, N, H*D) layout (== reshape(B,N,C)).
#pragma unroll
  for (int jd = 0; jd < 4; ++jd)
#pragma unroll
    for (int r = 0; r < 8; ++r) {
      const float denom = lrow[r];
      const float val   = (denom > 0.f) ? o[jd][r] / denom : 0.f;
      const size_t idx =
          ((size_t)b * Nn + n0 + r + 8 * hi) * Cc + h * 64 + jd * 16 + l15;
      outO[idx] = (__bf16)val;
    }
}

// ---------------------------------------------------------------------------
// Host orchestration
// ---------------------------------------------------------------------------
extern "C" void kernel_launch(void* const* d_in, const int* in_sizes, int n_in,
                              void* d_out, int out_size, void* d_ws, size_t ws_size,
                              hipStream_t stream) {
  (void)in_sizes; (void)n_in; (void)out_size; (void)ws_size;
  constexpr int L_ = 4, B_ = 2, N_ = 1024, C_ = 1024, H_ = 16, D_ = 64, F_ = 4096;
  constexpr int MN = B_ * N_;  // 2048 rows of the residual stream

  const float* x    = (const float*)d_in[0];
  const unsigned char* amask = (const unsigned char*)d_in[1];  // jnp bool = 1 byte
  const float* pos  = (const float*)d_in[2];
  const float* Wq   = (const float*)d_in[3];
  const float* bq   = (const float*)d_in[4];
  const float* Wk   = (const float*)d_in[5];
  const float* bk   = (const float*)d_in[6];
  const float* Wv   = (const float*)d_in[7];
  const float* bv   = (const float*)d_in[8];
  const float* Wo   = (const float*)d_in[9];
  const float* bo   = (const float*)d_in[10];
  const float* Wp   = (const float*)d_in[11];
  const float* pbu  = (const float*)d_in[12];
  const float* pbv  = (const float*)d_in[13];
  const float* g1   = (const float*)d_in[14];
  const float* be1  = (const float*)d_in[15];
  const float* g2   = (const float*)d_in[16];
  const float* be2  = (const float*)d_in[17];
  const float* W1   = (const float*)d_in[18];
  const float* bf1  = (const float*)d_in[19];
  const float* W2   = (const float*)d_in[20];
  const float* bf2  = (const float*)d_in[21];
  const float* gf   = (const float*)d_in[22];
  const float* betaf= (const float*)d_in[23];
  float* out = (float*)d_out;

  // --- workspace carve-up (256B aligned slices) ---
  char* ws = (char*)d_ws;
  size_t off = 0;
  auto alloc = [&](size_t bytes) -> char* {
    char* p = ws + off;
    off += (bytes + 255) & ~(size_t)255;
    return p;
  };
  float*  hA   = (float*)alloc((size_t)MN * C_ * 4);   // residual stream (ping)
  float*  hB   = (float*)alloc((size_t)MN * C_ * 4);   // residual stream (pong)
  __bf16* hnb  = (__bf16*)alloc((size_t)MN * C_ * 2);  // LN output (bf16)
  float*  qf   = (float*)alloc((size_t)MN * C_ * 4);   // q projection (f32)
  __bf16* qub  = (__bf16*)alloc((size_t)MN * C_ * 2);
  __bf16* qvb  = (__bf16*)alloc((size_t)MN * C_ * 2);
  __bf16* kb_  = (__bf16*)alloc((size_t)MN * C_ * 2);
  __bf16* vb_  = (__bf16*)alloc((size_t)MN * C_ * 2);
  __bf16* ob_  = (__bf16*)alloc((size_t)MN * C_ * 2);  // attention output
  __bf16* peb  = (__bf16*)alloc((size_t)N_ * C_ * 2);  // pos embeds bf16
  __bf16* pb_  = (__bf16*)alloc((size_t)N_ * C_ * 2);  // p projection bf16
  __bf16* f1b  = (__bf16*)alloc((size_t)MN * F_ * 2);  // FFN mid
  __bf16* wqb  = (__bf16*)alloc((size_t)C_ * C_ * 2);
  __bf16* wkb  = (__bf16*)alloc((size_t)C_ * C_ * 2);
  __bf16* wvb  = (__bf16*)alloc((size_t)C_ * C_ * 2);
  __bf16* wob  = (__bf16*)alloc((size_t)C_ * C_ * 2);
  __bf16* wpb  = (__bf16*)alloc((size_t)C_ * C_ * 2);
  __bf16* w1b  = (__bf16*)alloc((size_t)C_ * F_ * 2);
  __bf16* w2b  = (__bf16*)alloc((size_t)F_ * C_ * 2);

  auto conv = [&](const float* s, __bf16* d, int n) {
    f32_to_bf16_kernel<<<(n / 4 + 255) / 256, 256, 0, stream>>>(s, d, n);
  };

  // h = x ; pos embeds -> bf16 once
  hipMemcpyAsync(hA, x, (size_t)MN * C_ * 4, hipMemcpyDeviceToDevice, stream);
  conv(pos, peb, N_ * C_);

  const dim3 gCC(C_ / 128, MN / 128);
  const dim3 gP(C_ / 128, N_ / 128);
  const dim3 gF1(F_ / 128, MN / 128);

  for (int i = 0; i < L_; ++i) {
    conv(Wq + (size_t)i * C_ * C_, wqb, C_ * C_);
    conv(Wk + (size_t)i * C_ * C_, wkb, C_ * C_);
    conv(Wv + (size_t)i * C_ * C_, wvb, C_ * C_);
    conv(Wo + (size_t)i * C_ * C_, wob, C_ * C_);
    conv(Wp + (size_t)i * C_ * C_, wpb, C_ * C_);
    conv(W1 + (size_t)i * C_ * F_, w1b, C_ * F_);
    conv(W2 + (size_t)i * F_ * C_, w2b, F_ * C_);

    // LN1 -> bf16
    layernorm_kernel<<<MN, 256, 0, stream>>>(hA, g1 + (size_t)i * C_,
                                             be1 + (size_t)i * C_,
                                             nullptr, hnb, C_);
    // Projections: q -> f32 (needs bias_u/v add), k/v/p -> bf16 directly
    gemm_bf16_wmma<true, false, false, true, false><<<gCC, 256, 0, stream>>>(
        hnb, wqb, bq + (size_t)i * C_, nullptr, qf, nullptr, MN, C_, C_);
    gemm_bf16_wmma<true, false, false, false, true><<<gCC, 256, 0, stream>>>(
        hnb, wkb, bk + (size_t)i * C_, nullptr, nullptr, kb_, MN, C_, C_);
    gemm_bf16_wmma<true, false, false, false, true><<<gCC, 256, 0, stream>>>(
        hnb, wvb, bv + (size_t)i * C_, nullptr, nullptr, vb_, MN, C_, C_);
    gemm_bf16_wmma<false, false, false, false, true><<<gP, 256, 0, stream>>>(
        peb, wpb, nullptr, nullptr, nullptr, pb_, N_, C_, C_);
    make_quv_kernel<<<(MN * C_ / 4 + 255) / 256, 256, 0, stream>>>(
        qf, pbu + (size_t)i * H_ * D_, pbv + (size_t)i * H_ * D_, qub, qvb,
        MN * C_);
    // Attention
    attention_kernel<<<B_ * H_ * (N_ / 128), 256, 0, stream>>>(
        qub, qvb, kb_, vb_, pb_, amask, ob_);
    // Output projection + residual -> hB
    gemm_bf16_wmma<true, false, true, true, false><<<gCC, 256, 0, stream>>>(
        ob_, wob, bo + (size_t)i * C_, hA, hB, nullptr, MN, C_, C_);
    // LN2 -> bf16
    layernorm_kernel<<<MN, 256, 0, stream>>>(hB, g2 + (size_t)i * C_,
                                             be2 + (size_t)i * C_,
                                             nullptr, hnb, C_);
    // FFN: relu(hn@W1 + bf1) @ W2 + bf2 + residual
    gemm_bf16_wmma<true, true, false, false, true><<<gF1, 256, 0, stream>>>(
        hnb, w1b, bf1 + (size_t)i * F_, nullptr, nullptr, f1b, MN, F_, C_);
    gemm_bf16_wmma<true, false, true, true, false><<<gCC, 256, 0, stream>>>(
        f1b, w2b, bf2 + (size_t)i * C_, hB, hA, nullptr, MN, C_, F_);
  }

  // Final LN -> f32 output
  layernorm_kernel<<<MN, 256, 0, stream>>>(hA, gf, betaf, out, nullptr, C_);
}
